// KNNModel_58540404244549
// MI455X (gfx1250) — matrix operand order repaired and checked by reading, not compiled, not run on hardware
//
#include <hip/hip_runtime.h>
#include <hip/hip_bf16.h>
#include <math.h>

typedef __attribute__((ext_vector_type(16))) _Float16 v16h;
typedef __attribute__((ext_vector_type(8)))  _Float16 v8h;
typedef __attribute__((ext_vector_type(8)))  float    v8f;
typedef __attribute__((ext_vector_type(4)))  float    v4f;
typedef __attribute__((ext_vector_type(4)))  unsigned v4u;
typedef __attribute__((ext_vector_type(8)))  int      v8i;
typedef __attribute__((ext_vector_type(4)))  int      v4i;

#define DIM       512
#define BQ        1024
#define NBASE     100000
#define NPAD      100352              // multiple of 16*NSPLIT = 1024
#define NSPLIT    64
#define COLS_PS   (NPAD / NSPLIT)     // 1568 columns per split
#define TILES_PS  (COLS_PS / 16)      // 98 16-col tiles per split
#define KTOP      5
#define MROWS     128                 // query rows per workgroup (16 per wave)
#define BSTRIDE   520                 // halves per row in LDS B buffer (pad 8)
#define BBUF_B    (16 * BSTRIDE * 2)  // 16640 bytes per B buffer

// ---- workspace layout (bytes) ----
#define WS_BH     0u
#define WS_XH     102760448u          // NPAD*DIM*2
#define WS_BNORM  103809024u          // + BQ*DIM*2
#define WS_XNORM  104210432u          // + NPAD*4
#define WS_PART   104214528u          // + 4096  (part: BQ*NSPLIT*KTOP floats)

union V16U { v16h v; v8h h[2]; };

__device__ __forceinline__ unsigned umin_(unsigned a, unsigned b) { return a < b ? a : b; }
__device__ __forceinline__ unsigned umax_(unsigned a, unsigned b) { return a > b ? a : b; }

// monotone IEEE-float -> uint32 key (ascending order preserved)
__device__ __forceinline__ unsigned fkey(float f) {
    unsigned u = __float_as_uint(f);
    return u ^ (0x80000000u | (unsigned)((int)u >> 31));
}
__device__ __forceinline__ float funkey(unsigned k) {
    unsigned u = (k & 0x80000000u) ? (k ^ 0x80000000u) : ~k;
    return __uint_as_float(u);
}

// branchless insert into ascending sorted 5-smallest key list (pure int min/max)
__device__ __forceinline__ void ins5u(unsigned (&t)[KTOP], unsigned v) {
    t[4] = umin_(t[4], v);
    unsigned x;
    x = umin_(t[3], t[4]); t[4] = umax_(t[3], t[4]); t[3] = x;
    x = umin_(t[2], t[3]); t[3] = umax_(t[2], t[3]); t[2] = x;
    x = umin_(t[1], t[2]); t[2] = umax_(t[1], t[2]); t[1] = x;
    x = umin_(t[0], t[1]); t[1] = umax_(t[0], t[1]); t[0] = x;
}

__device__ __forceinline__ void ins5f(float (&t)[KTOP], float v) {
    t[4] = fminf(t[4], v);
    float x;
    x = fminf(t[3], t[4]); t[4] = fmaxf(t[3], t[4]); t[3] = x;
    x = fminf(t[2], t[3]); t[3] = fmaxf(t[2], t[3]); t[2] = x;
    x = fminf(t[1], t[2]); t[2] = fmaxf(t[1], t[2]); t[1] = x;
    x = fminf(t[0], t[1]); t[1] = fmaxf(t[0], t[1]); t[0] = x;
}

// TDM: DMA one 16-row x 512-half B tile (16KB) global -> LDS, with row padding
// of 4 dwords after every 256 dwords (row stride becomes 520 halves in LDS).
// D# layout per CDNA5 ISA ch.8 (group0 128b, group1 256b; groups 2/3 unused).
__device__ __forceinline__ void tdm_load_tile(unsigned lds_off, const _Float16* gsrc) {
    unsigned long long ga = (unsigned long long)(uintptr_t)gsrc;
    v4u g0;
    g0[0] = 1u;                                   // count=1, user descriptor
    g0[1] = lds_off;                              // lds_addr (bytes)
    g0[2] = (unsigned)(ga & 0xFFFFFFFFu);         // global_addr[31:0]
    g0[3] = (unsigned)((ga >> 32) & 0x01FFFFFFu)  // global_addr[56:32]
          | (2u << 30);                           // type = 2 ("image")
    v8i g1;
    g1[0] = (1 << 16)                             // data_size = 1 -> 2 bytes
          | (1 << 20)                             // pad_enable
          | (7 << 22)                             // pad_interval: 256 dwords
          | (3 << 25);                            // pad_amount: 4 dwords (16B)
    g1[1] = (512 << 16);                          // tensor_dim0 = 512
    g1[2] = (16 << 16);                           // tensor_dim1 = 16
    g1[3] = (512 << 16);                          // tile_dim0 = 512
    g1[4] = 16;                                   // tile_dim1 = 16 (tile_dim2=0)
    g1[5] = 512;                                  // tensor_dim0_stride = 512
    g1[6] = 0;
    g1[7] = 0;
    v4i gz4 = {0, 0, 0, 0};
    v8i gz8 = {0, 0, 0, 0, 0, 0, 0, 0};
    __builtin_amdgcn_tensor_load_to_lds(g0, g1, gz4, gz4, gz8, 0);
}

// fp32 rows -> f16 rows + fp32 row norms; pad rows zero-filled, pad norm = +INF
__global__ void knn_convert_rows(const float* __restrict__ src,
                                 _Float16* __restrict__ dst,
                                 float* __restrict__ norms,
                                 int nvalid, int ntotal) {
    int wave = threadIdx.x >> 5, lane = threadIdx.x & 31;
    int row  = blockIdx.x * 8 + wave;
    if (row >= ntotal) return;
    _Float16* drow = dst + (size_t)row * DIM;
    if (row < nvalid) {
        const float* srow = src + (size_t)row * DIM;
        const v4f* s4 = ((const v4f*)srow) + lane * 4;   // 16 floats per lane
        float acc = 0.0f;
        v8h o0, o1;
        #pragma unroll
        for (int i = 0; i < 4; ++i) {
            v4f v = s4[i];
            #pragma unroll
            for (int j = 0; j < 4; ++j) {
                float f = v[j];
                acc = fmaf(f, f, acc);
                int idx = i * 4 + j;
                if (idx < 8) o0[idx] = (_Float16)f; else o1[idx - 8] = (_Float16)f;
            }
        }
        *(v8h*)(drow + lane * 16)     = o0;
        *(v8h*)(drow + lane * 16 + 8) = o1;
        #pragma unroll
        for (int off = 16; off >= 1; off >>= 1) acc += __shfl_xor(acc, off, 32);
        if (lane == 0) norms[row] = acc;
    } else {
        v8h z;
        #pragma unroll
        for (int j = 0; j < 8; ++j) z[j] = (_Float16)0.0f;
        *(v8h*)(drow + lane * 16)     = z;
        *(v8h*)(drow + lane * 16 + 8) = z;
        if (lane == 0) norms[row] = __builtin_inff();   // never selected
    }
}

union __align__(16) SmemT {
    _Float16 b[3][16 * BSTRIDE];            // triple-buffered B tiles (49920 B)
    unsigned mg[8 * 32 * 8 * KTOP];         // within-wave merge buffer (40960 B)
};

// fused GEMM (f16 WMMA, TDM-staged B) + per-query running top-5
__global__ void __launch_bounds__(256)
knn_main(const _Float16* __restrict__ xh,
         const _Float16* __restrict__ bh,
         const float*    __restrict__ bnorm,
         float*          __restrict__ part) {
    __shared__ SmemT sm;
    const int tid   = threadIdx.x;
    const int mt    = blockIdx.x;     // query tile (MROWS rows)
    const int split = blockIdx.y;     // N split
    const int wave  = tid >> 5, lane = tid & 31;
    const int row   = lane & 15, sel = lane >> 4;
    const int colbase = split * COLS_PS;

    // preload this wave's 16 A fragments (16 rows x 512 K) into registers
    V16U Af[16];
    {
        const _Float16* arow =
            xh + (size_t)(mt * MROWS + wave * 16 + row) * DIM + sel * 8;
        #pragma unroll
        for (int kk = 0; kk < 16; ++kk) {
            Af[kk].h[0] = *(const v8h*)(arow + kk * 32);
            Af[kk].h[1] = *(const v8h*)(arow + kk * 32 + 16);
        }
    }

    unsigned top[8][KTOP];
    #pragma unroll
    for (int r = 0; r < 8; ++r)
        #pragma unroll
        for (int j = 0; j < KTOP; ++j) top[r][j] = 0xFFFFFFFFu;

    const unsigned lds0 = (unsigned)(uintptr_t)&sm.b[0][0];

    // prime the TDM pipeline: tiles 0 and 1
    if (wave == 0) {
        tdm_load_tile(lds0,          bh + (size_t)(colbase +  0) * DIM);
        tdm_load_tile(lds0 + BBUF_B, bh + (size_t)(colbase + 16) * DIM);
    }

    int cur = 0;
    for (int i = 0; i < TILES_PS; ++i) {
        if (wave == 0) {
            if (i + 1 < TILES_PS) __builtin_amdgcn_s_wait_tensorcnt(1);
            else                  __builtin_amdgcn_s_wait_tensorcnt(0);
        }
        __syncthreads();   // tile i staged for all; all done reading tile i-1's buf
        if (wave == 0 && i + 2 < TILES_PS) {
            int nb = cur - 1; if (nb < 0) nb += 3;         // (cur+2)%3
            tdm_load_tile(lds0 + (unsigned)nb * BBUF_B,
                          bh + (size_t)(colbase + (i + 2) * 16) * DIM);
        }

        const int col0 = colbase + i * 16;
        const float bn = bnorm[col0 + row];

        v8f acc;
        #pragma unroll
        for (int c = 0; c < 8; ++c) acc[c] = 0.0f;

        // 2-buffer LDS-read ring, prefetch distance 2:
        // preload frags 0,1; after wmma(kk) load frag kk+2 into the freed buffer
        const _Float16* brow = &sm.b[cur][0] + row * BSTRIDE + sel * 8;
        V16U Bb[2];
        #pragma unroll
        for (int p = 0; p < 2; ++p) {
            const _Float16* bp = brow + p * 32;
            Bb[p].h[0] = *(const v8h*)bp;
            Bb[p].h[1] = *(const v8h*)(bp + 16);
        }
        #pragma unroll
        for (int kk = 0; kk < 16; ++kk) {
            acc = __builtin_amdgcn_wmma_f32_16x16x32_f16(
                false, Af[kk].v, false, Bb[kk & 1].v, (short)0, acc, false, false);
            if (kk < 14) {
                const _Float16* bp = brow + (kk + 2) * 32;
                Bb[kk & 1].h[0] = *(const v8h*)bp;
                Bb[kk & 1].h[1] = *(const v8h*)(bp + 16);
            }
        }
        #pragma unroll
        for (int r = 0; r < 8; ++r)
            ins5u(top[r], fkey(fmaf(-2.0f, acc[r], bn)));

        cur = (cur + 1 == 3) ? 0 : cur + 1;
    }

    __syncthreads();   // done with B buffers; reuse LDS as merge buffer
    #pragma unroll
    for (int r = 0; r < 8; ++r)
        #pragma unroll
        for (int j = 0; j < KTOP; ++j)
            sm.mg[(((wave << 5) | lane) * 8 + r) * KTOP + j] = top[r][j];
    __syncthreads();

    if (tid < MROWS) {
        const int m  = tid;               // local query row 0..127
        const int w  = m >> 4;            // owning wave
        const int laneBase = ((m >> 3) & 1) * 16;
        const int rr = m & 7;
        unsigned best[KTOP];
        #pragma unroll
        for (int j = 0; j < KTOP; ++j) best[j] = 0xFFFFFFFFu;
        for (int l = 0; l < 16; ++l) {
            const unsigned* src = &sm.mg[((w * 32 + laneBase + l) * 8 + rr) * KTOP];
            #pragma unroll
            for (int j = 0; j < KTOP; ++j) ins5u(best, src[j]);
        }
        float* dst = part + ((size_t)(mt * MROWS + m) * NSPLIT + split) * KTOP;
        #pragma unroll
        for (int j = 0; j < KTOP; ++j) dst[j] = funkey(best[j]);
    }
}

// merge NSPLIT partial top-5 lists per query, apply +||x||^2 and sqrt
__global__ void knn_merge(const float* __restrict__ part,
                          const float* __restrict__ xnorm,
                          float* __restrict__ out) {
    int q = blockIdx.x * blockDim.x + threadIdx.x;
    if (q >= BQ) return;
    float best[KTOP];
    #pragma unroll
    for (int j = 0; j < KTOP; ++j) best[j] = 3.0e38f;
    const float* src = part + (size_t)q * NSPLIT * KTOP;
    for (int i = 0; i < NSPLIT * KTOP; ++i) ins5f(best, src[i]);
    const float xn = xnorm[q];
    #pragma unroll
    for (int j = 0; j < KTOP; ++j)
        out[q * KTOP + j] = sqrtf(fmaxf(xn + best[j], 0.0f));
}

extern "C" void kernel_launch(void* const* d_in, const int* in_sizes, int n_in,
                              void* d_out, int out_size, void* d_ws, size_t ws_size,
                              hipStream_t stream) {
    const float* x    = (const float*)d_in[0];   // [1024,512] fp32
    const float* base = (const float*)d_in[1];   // [100000,512] fp32
    (void)in_sizes; (void)n_in; (void)out_size; (void)ws_size;

    char* ws = (char*)d_ws;
    _Float16* bh    = (_Float16*)(ws + WS_BH);
    _Float16* xh    = (_Float16*)(ws + WS_XH);
    float*    bnorm = (float*)   (ws + WS_BNORM);
    float*    xnorm = (float*)   (ws + WS_XNORM);
    float*    part  = (float*)   (ws + WS_PART);
    float*    out   = (float*)d_out;

    knn_convert_rows<<<NPAD / 8, 256, 0, stream>>>(base, bh, bnorm, NBASE, NPAD);
    knn_convert_rows<<<BQ / 8,   256, 0, stream>>>(x,    xh, xnorm, BQ,    BQ);
    knn_main<<<dim3(BQ / MROWS, NSPLIT), 256, 0, stream>>>(xh, bh, bnorm, part);
    knn_merge<<<(BQ + 255) / 256, 256, 0, stream>>>(part, xnorm, out);
}
